// ProteinFeatures_25211458027662
// MI455X (gfx1250) — matrix operand order repaired and checked by reading, not compiled, not run on hardware
//
#include <hip/hip_runtime.h>
#include <hip/hip_bf16.h>

typedef __attribute__((ext_vector_type(16))) _Float16 v16h;
typedef __attribute__((ext_vector_type(8)))  _Float16 v8h;
typedef __attribute__((ext_vector_type(8)))  float    v8f;

#define TOPK 30
#define LMAXROW 2048
#define EDGE_IN 416   // 16 pos-emb + 25*16 RBF
#define EDGE_OUT 128

__constant__ int cPA[25] = {0,1,2,3,4,0,0,0,0,1,1,1,4,4,3,1,2,3,4,2,3,4,2,3,2};
__constant__ int cPB[25] = {0,1,2,3,4,1,2,3,4,2,3,4,2,3,2,0,0,0,0,1,1,1,4,4,3};

// ---------------------------------------------------------------------------
// Kernel 1: build 5 atoms per residue: [Ca, N, C, O, Cb] as float3 each
// ---------------------------------------------------------------------------
__global__ void k_atoms(const float* __restrict__ X, float* __restrict__ atoms, int BL) {
    int t = blockIdx.x * blockDim.x + threadIdx.x;
    if (t >= BL) return;
    const float* x = X + (size_t)t * 12;   // 4 atoms x 3
    float Nx = x[0], Ny = x[1],  Nz = x[2];
    float Ax = x[3], Ay = x[4],  Az = x[5];   // Ca
    float Cx = x[6], Cy = x[7],  Cz = x[8];
    float Ox = x[9], Oy = x[10], Oz = x[11];
    float bx = Ax - Nx, by = Ay - Ny, bz = Az - Nz;
    float cx = Cx - Ax, cy = Cy - Ay, cz = Cz - Az;
    // a = cross(b, c)
    float ax = by * cz - bz * cy;
    float ay = bz * cx - bx * cz;
    float az = bx * cy - by * cx;
    float Bx = -0.58273431f * ax + 0.56802827f * bx - 0.54067466f * cx + Ax;
    float By = -0.58273431f * ay + 0.56802827f * by - 0.54067466f * cy + Ay;
    float Bz = -0.58273431f * az + 0.56802827f * bz - 0.54067466f * cz + Az;
    float* o = atoms + (size_t)t * 15;
    o[0]=Ax; o[1]=Ay; o[2]=Az;      // Ca
    o[3]=Nx; o[4]=Ny; o[5]=Nz;      // N
    o[6]=Cx; o[7]=Cy; o[8]=Cz;      // C
    o[9]=Ox; o[10]=Oy; o[11]=Oz;    // O
    o[12]=Bx; o[13]=By; o[14]=Bz;   // Cb
}

// ---------------------------------------------------------------------------
// Kernel 2: per (b,i) row: masked Ca distances to all j, top-30 smallest.
// One block (256 threads) per row; D row lives in LDS.
// ---------------------------------------------------------------------------
__global__ __launch_bounds__(256) void k_topk(const float* __restrict__ atoms,
                                              const float* __restrict__ mask,
                                              int* __restrict__ eidx, int L, int K) {
    __shared__ float Dn[LMAXROW];
    __shared__ float rv[256];
    __shared__ int   ri[256];
    int row = blockIdx.x;            // b*L + i
    int b   = row / L;
    int base = b * L;
    int tid = threadIdx.x;
    float cax = atoms[(size_t)row*15+0];
    float cay = atoms[(size_t)row*15+1];
    float caz = atoms[(size_t)row*15+2];
    float mi  = mask[row];
    float lmax = 0.0f;
    for (int j = tid; j < L; j += 256) {
        const float* aj = atoms + (size_t)(base + j) * 15;
        float dx = cax - aj[0], dy = cay - aj[1], dz = caz - aj[2];
        float m2 = mi * mask[base + j];
        float d  = m2 * sqrtf(dx*dx + dy*dy + dz*dz + 1e-6f);
        Dn[j] = d;
        lmax = fmaxf(lmax, d);
    }
    rv[tid] = lmax;
    __syncthreads();
    for (int s = 128; s > 0; s >>= 1) {
        if (tid < s) rv[tid] = fmaxf(rv[tid], rv[tid + s]);
        __syncthreads();
    }
    float dmax = rv[0];
    __syncthreads();
    for (int j = tid; j < L; j += 256) {
        float m2 = mi * mask[base + j];
        Dn[j] += (1.0f - m2) * dmax;
    }
    __syncthreads();
    for (int kk = 0; kk < K; ++kk) {
        float bv = 3.4e38f; int bi = 0x7fffffff;
        for (int j = tid; j < L; j += 256) {
            float v = Dn[j];
            if (v < bv || (v == bv && j < bi)) { bv = v; bi = j; }
        }
        rv[tid] = bv; ri[tid] = bi;
        __syncthreads();
        for (int s = 128; s > 0; s >>= 1) {
            if (tid < s) {
                if (rv[tid+s] < rv[tid] || (rv[tid+s] == rv[tid] && ri[tid+s] < ri[tid])) {
                    rv[tid] = rv[tid+s]; ri[tid] = ri[tid+s];
                }
            }
            __syncthreads();
        }
        if (tid == 0) { eidx[(size_t)row*K + kk] = ri[0]; Dn[ri[0]] = 3.4e38f; }
        __syncthreads();
    }
}

// ---------------------------------------------------------------------------
// Kernel 3: W_edge (416x128 f32, K-major) -> Wt (128x416 f16, N-major)
// so that each WMMA B fragment is one contiguous aligned 32B load per lane.
// ---------------------------------------------------------------------------
__global__ void k_wt(const float* __restrict__ W_edge, _Float16* __restrict__ Wt) {
    int t = blockIdx.x * blockDim.x + threadIdx.x;
    if (t >= EDGE_IN * EDGE_OUT) return;
    int k = t / EDGE_OUT;
    int n = t % EDGE_OUT;
    Wt[(size_t)n * EDGE_IN + k] = (_Float16)W_edge[t];
}

// ---------------------------------------------------------------------------
// Kernel 4 (fused): per block = 16 edges.
//   Phase A: build 16x416 f16 feature tile in LDS (pos-emb + 25x16 RBF)
//   Phase B: 8 waves x 13 x v_wmma_f32_16x16x32_f16 -> 16x128 f32
//   Phase C: LayerNorm(128) via LDS + wave32 shuffles, write to d_out
// ---------------------------------------------------------------------------
__global__ __launch_bounds__(256) void k_edge(const float* __restrict__ atoms,
                                              const int* __restrict__ eidx,
                                              const int* __restrict__ resid,
                                              const float* __restrict__ W_pos,
                                              const float* __restrict__ b_pos,
                                              const _Float16* __restrict__ Wt,
                                              const float* __restrict__ gamma,
                                              const float* __restrict__ beta,
                                              float* __restrict__ out,
                                              int L, int K) {
    __shared__ __align__(32) _Float16 Af[16 * EDGE_IN];   // 13312 B
    __shared__ float Ct[16 * EDGE_OUT];                   //  8192 B
    __shared__ int sRow[16];   // global residue row b*L+i per edge
    __shared__ int sJ[16];     // neighbor j (per-batch index)

    int tid = threadIdx.x;
    long e0 = (long)blockIdx.x * 16;

    if (tid < 16) {
        long e = e0 + tid;
        sRow[tid] = (int)(e / K);
        sJ[tid]   = eidx[e];
    }
    __syncthreads();

    // --- positional embedding: one-hot @ W_pos == row gather, + b_pos ---
    {
        int r = tid >> 4, c = tid & 15;
        int row = sRow[r];
        int b   = row / L;
        int j   = sJ[r];
        int off = resid[row] - resid[b * L + j];
        int d = off + 32;
        d = d < 0 ? 0 : (d > 64 ? 64 : d);
        Af[r * EDGE_IN + c] = (_Float16)(W_pos[d * 16 + c] + b_pos[c]);
    }
    // --- RBF features: 16 edges x 25 atom pairs ---
    for (int task = tid; task < 400; task += 256) {
        int r = task / 25, p = task % 25;
        int row = sRow[r];
        int b   = row / L;
        int j   = sJ[r];
        const float* pa = atoms + (size_t)row * 15            + cPA[p] * 3;
        const float* pb = atoms + (size_t)(b * L + j) * 15    + cPB[p] * 3;
        float dx = pa[0]-pb[0], dy = pa[1]-pb[1], dz = pa[2]-pb[2];
        float dist = sqrtf(dx*dx + dy*dy + dz*dz + 1e-6f);
        _Float16* fp = Af + r * EDGE_IN + 16 + p * 16;
#pragma unroll
        for (int rb = 0; rb < 16; ++rb) {
            float mu = 2.0f + (20.0f / 15.0f) * rb;       // linspace(2,22,16)
            float z  = (dist - mu) * 0.8f;                 // sigma = 1.25
            fp[rb] = (_Float16)__expf(-z * z);
        }
    }
    __syncthreads();

    // --- WMMA: each wave owns one 16-wide N slice ---
    int wave = tid >> 5;
    int lane = tid & 31;
    int mrow = lane & 15;
    int sel  = (lane & 16) ? 16 : 0;      // A: K half per half-wave
    int koff = (lane & 16) ? 16 : 0;      // B: K half per half-wave
    int ncol = (wave << 4) + (lane & 15);
    const _Float16* wbase = Wt + (size_t)ncol * EDGE_IN + koff;

    v8f acc = {};
#pragma unroll
    for (int kt = 0; kt < 13; ++kt) {
        int k0 = kt * 32;
        union { v16h v; v8h h[2]; } A;
        A.h[0] = *(const v8h*)(Af + mrow * EDGE_IN + k0 + sel);
        A.h[1] = *(const v8h*)(Af + mrow * EDGE_IN + k0 + sel + 8);
        v16h Bv = *(const v16h*)(wbase + k0);
        if (kt + 1 < 13) __builtin_prefetch(wbase + k0 + 32, 0, 1);
        acc = __builtin_amdgcn_wmma_f32_16x16x32_f16(
            /*neg_a=*/false, A.v, /*neg_b=*/false, Bv,
            /*c_mod=*/(short)0, acc, /*reuse_a=*/false, /*reuse_b=*/false);
    }

    // --- scatter C fragment to LDS (C layout: VGPR v -> M=v / v+8) ---
    int crow = (lane & 16) ? 8 : 0;
#pragma unroll
    for (int v = 0; v < 8; ++v)
        Ct[(crow + v) * EDGE_OUT + ncol] = acc[v];
    __syncthreads();

    // --- LayerNorm over 128: each half-wave handles one of 16 rows ---
    int r  = (wave << 1) + ((lane >> 4) & 1);
    int cl = lane & 15;
    float s = 0.0f, ss = 0.0f;
    float vals[8];
#pragma unroll
    for (int q = 0; q < 8; ++q) {
        float x = Ct[r * EDGE_OUT + cl + q * 16];
        vals[q] = x; s += x; ss += x * x;
    }
    s  += __shfl_xor(s, 1, 16);  ss += __shfl_xor(ss, 1, 16);
    s  += __shfl_xor(s, 2, 16);  ss += __shfl_xor(ss, 2, 16);
    s  += __shfl_xor(s, 4, 16);  ss += __shfl_xor(ss, 4, 16);
    s  += __shfl_xor(s, 8, 16);  ss += __shfl_xor(ss, 8, 16);
    float mean = s * (1.0f / 128.0f);
    float var  = ss * (1.0f / 128.0f) - mean * mean;
    float inv  = rsqrtf(var + 1e-5f);
    size_t obase = (size_t)(e0 + r) * EDGE_OUT;
#pragma unroll
    for (int q = 0; q < 8; ++q) {
        int c = cl + q * 16;
        out[obase + c] = (vals[q] - mean) * inv * gamma[c] + beta[c];
    }
}

// ---------------------------------------------------------------------------
extern "C" void kernel_launch(void* const* d_in, const int* in_sizes, int n_in,
                              void* d_out, int out_size, void* d_ws, size_t ws_size,
                              hipStream_t stream) {
    const float* X      = (const float*)d_in[0];
    const float* mask   = (const float*)d_in[1];
    const int*   resid  = (const int*)d_in[2];
    const float* W_pos  = (const float*)d_in[3];
    const float* b_pos  = (const float*)d_in[4];
    const float* W_edge = (const float*)d_in[5];
    const float* gamma  = (const float*)d_in[6];
    const float* beta   = (const float*)d_in[7];
    float* out = (float*)d_out;

    int BL = in_sizes[1];                 // B * L (mask element count)
    int L  = (BL >= LMAXROW) ? LMAXROW : BL;
    int K  = TOPK;

    // workspace layout
    char* ws = (char*)d_ws;
    float*    atoms = (float*)ws;                                   // BL*15 f32
    int*      eidx  = (int*)(ws + (size_t)BL * 15 * sizeof(float)); // BL*K  i32
    _Float16* Wt    = (_Float16*)(ws + (size_t)BL * 15 * sizeof(float)
                                     + (size_t)BL * K  * sizeof(int)); // 128*416 f16

    k_atoms<<<(BL + 255) / 256, 256, 0, stream>>>(X, atoms, BL);
    k_topk <<<BL, 256, 0, stream>>>(atoms, mask, eidx, L, K);
    k_wt   <<<(EDGE_IN * EDGE_OUT + 255) / 256, 256, 0, stream>>>(W_edge, Wt);

    long M = (long)BL * K;                // 245760 edges, divisible by 16
    int blocks = (int)(M / 16);
    k_edge<<<blocks, 256, 0, stream>>>(atoms, eidx, resid, W_pos, b_pos, Wt,
                                       gamma, beta, out, L, K);
}
